// RNN_64381559767338
// MI455X (gfx1250) — matrix operand order repaired and checked by reading, not compiled, not run on hardware
//
#include <hip/hip_runtime.h>

// Problem constants (must match reference)
#define BB 64
#define TT 2048
#define II 128
#define HH 512
#define OO 64

__constant__ __device__ const float K_SIGMA_IN = 0.01f;
__constant__ __device__ const float K_SIGMA_RE = 0.01f;
__constant__ __device__ const float K_ALPHA    = 0.1f;   // DT/TAU = 10/100

typedef __attribute__((ext_vector_type(16))) _Float16 v16h;
typedef __attribute__((ext_vector_type(8)))  _Float16 v8h;
typedef __attribute__((ext_vector_type(4)))  _Float16 v4h;
typedef __attribute__((ext_vector_type(8)))  float    v8f;
typedef __attribute__((ext_vector_type(4)))  float    v4f;

union V16 { v16h v; v8h h[2]; };

// CDNA5 hardware tanh (TRANS32 op). s_delay_alu covers the required 1-slot
// delay before the TRANS result may be consumed (compiler can't see into asm).
__device__ __forceinline__ float htanh(float x) {
  float y;
  asm("v_tanh_f32 %0, %1\n\t"
      "s_delay_alu instid0(TRANS32_DEP_1)"
      : "=v"(y) : "v"(x));
  return y;
}

// ---------------------------------------------------------------------------
// WMMA-native tiled weight layout.
// A (32-K x 64-N) block of W^T is stored as: [chunk c: 8][lane: 32][8 x f16],
// i.e. element (k = 16(c&1) + 8(lane/16) + j, n = 16(c/2) + lane%16) at offset c*256 + lane*8 + j (f16 units).
// A lane's WMMA B-operand chunks are then at  base + lane*16B + c*512B  ->
// global_load_b128 with immediate offsets, 512 contiguous bytes per load
// across the wave.
// ---------------------------------------------------------------------------
#define BLK 2048  // f16 elements per (32K x 64N) block

__global__ void rnn_prep_weights(const float* __restrict__ Wih,
                                 const float* __restrict__ Whh,
                                 const float* __restrict__ Who,
                                 _Float16* __restrict__ WihTl,
                                 _Float16* __restrict__ WhhTl,
                                 _Float16* __restrict__ WhoTl) {
  const int i = blockIdx.x * blockDim.x + threadIdx.x;
  const int stride = gridDim.x * blockDim.x;
  // Whh: blk = kbb*8 + g
  for (int e = i; e < (HH / 32) * (HH / 64) * BLK; e += stride) {
    int blk = e / BLK, w = e % BLK;
    int c = w >> 8, lane = (w >> 3) & 31, j = w & 7;
    int kbb = blk >> 3, g = blk & 7;
    int k = kbb * 32 + (c & 1) * 16 + ((lane >> 4) << 3) + j, n = g * 64 + (c >> 1) * 16 + (lane & 15);
    WhhTl[e] = (_Float16)Whh[n * HH + k];
  }
  // Wih: blk = kbb*8 + g, kbb < 4
  for (int e = i; e < (II / 32) * (HH / 64) * BLK; e += stride) {
    int blk = e / BLK, w = e % BLK;
    int c = w >> 8, lane = (w >> 3) & 31, j = w & 7;
    int kbb = blk >> 3, g = blk & 7;
    int k = kbb * 32 + (c & 1) * 16 + ((lane >> 4) << 3) + j, n = g * 64 + (c >> 1) * 16 + (lane & 15);
    WihTl[e] = (_Float16)Wih[n * II + k];
  }
  // Who: blk = kbb, single 64-wide N group
  for (int e = i; e < (HH / 32) * BLK; e += stride) {
    int blk = e / BLK, w = e % BLK;
    int c = w >> 8, lane = (w >> 3) & 31, j = w & 7;
    int k = blk * 32 + (c & 1) * 16 + ((lane >> 4) << 3) + j, o = (c >> 1) * 16 + (lane & 15);
    WhoTl[e] = (_Float16)Who[o * HH + k];
  }
}

// ---------------------------------------------------------------------------
// Persistent RNN kernel: 4 blocks (one 16-row batch tile each), 256 threads
// = 8 waves. Each wave owns 64 columns of H (4 WMMA N-tiles). h (f32) and
// r = tanh(h) (f16) live in LDS; weights stream from L2 each step.
// ---------------------------------------------------------------------------
__global__ __launch_bounds__(256)
void rnn_persistent(const float* __restrict__ inputs,       // [B,T,I]
                    const float* __restrict__ hidden,       // [B,H]
                    const float* __restrict__ input_noise,  // [B,T,I]
                    const float* __restrict__ re_noise,     // [B,T,H]
                    const _Float16* __restrict__ WhhTl,     // tiled
                    const _Float16* __restrict__ WihTl,     // tiled
                    const _Float16* __restrict__ WhoTl,     // tiled
                    const float* __restrict__ bhh,          // [H]
                    const float* __restrict__ bho,          // [O]
                    float* __restrict__ hfinal,             // [B,H]
                    float* __restrict__ outs) {             // [B,T,O]
  // Row padding: 16 A-rows hit distinct LDS bank groups; rows stay 16B aligned.
  __shared__ float    sh_h[16][HH + 1];   // 32.8 KB
  __shared__ _Float16 sh_r[16][HH + 8];   // 16.6 KB
  __shared__ _Float16 sh_x[16][II + 8];   //  4.3 KB

  const int tid   = threadIdx.x;
  const int wave  = tid >> 5;
  const int lane  = tid & 31;
  const int bbase = blockIdx.x * 16;      // batch tile of 16 rows

  // WMMA A-operand addressing (16-bit A 16x32 layout).
  const int mA    = lane & 15;
  const int koff  = (lane >> 4) << 3;     // 0 or 8
  // C/D 16x16 f32 layout: VGPR v -> M = v (+8 for lanes 16-31), N = lane%16.
  const int mrow  = (lane >> 4) << 3;
  const int nlane = lane & 15;
  const int lane8 = lane << 3;            // lane's chunk base within a block

  // Init h from `hidden`, r = tanh(h).
  for (int e = tid; e < 16 * HH; e += 256) {
    int m = e >> 9, n = e & (HH - 1);
    float h0 = hidden[(bbase + m) * HH + n];
    sh_h[m][n] = h0;
    sh_r[m][n] = (_Float16)htanh(h0);
  }
  __syncthreads();

  for (int t = 0; t < TT; ++t) {
    // Launder a zero SGPR offset: weight addresses become loop-variant so
    // LICM cannot hoist weight loads out of the t-loop (avoids VGPR spill),
    // while the base pointers keep kernel-arg provenance -> global_load.
    int lz = 0;
    asm volatile("" : "+s"(lz));
    const _Float16* __restrict__ whh = WhhTl + lz;
    const _Float16* __restrict__ wih = WihTl + lz;
    const _Float16* __restrict__ who = WhoTl + lz;

    // ---- phase 1: x_tilde = x * (1 + sigma_in * n_in) into LDS (f16) ----
    // float4 loads, 8-byte LDS stores.
    for (int e = tid; e < 16 * (II / 4); e += 256) {
      int m = e >> 5, k4 = (e & 31) << 2;
      int g = ((bbase + m) * TT + t) * II + k4;
      v4f xv = *(const v4f*)(inputs + g);
      v4f nv = *(const v4f*)(input_noise + g);
      v4h xo;
      #pragma unroll
      for (int j = 0; j < 4; ++j)
        xo[j] = (_Float16)(xv[j] * (1.0f + K_SIGMA_IN * nv[j]));
      *(v4h*)&sh_x[m][k4] = xo;
      if (t + 1 < TT) {                         // warm L2 for next step
        __builtin_prefetch(inputs + g + II, 0, 1);
        __builtin_prefetch(input_noise + g + II, 0, 1);
      }
    }
    __syncthreads();

    // ---- phase 2: acc = r @ Whh^T + x~ @ Wih^T for this wave's 64 columns --
    // One A-tile (2 ds_load_b128) feeds 4 WMMAs; B chunks are imm-offset loads.
    v8f acc[4];
    #pragma unroll
    for (int nt = 0; nt < 4; ++nt) acc[nt] = (v8f){0,0,0,0,0,0,0,0};

    #pragma unroll 1
    for (int kbb = 0; kbb < HH / 32; ++kbb) {
      const int kb = kbb * 32;
      V16 a;
      a.h[0] = *(const v8h*)&sh_r[mA][kb + koff];
      a.h[1] = *(const v8h*)&sh_r[mA][kb + 16 + koff];
      const _Float16* base = whh + (kbb * 8 + wave) * BLK + lane8;
      #pragma unroll
      for (int nt = 0; nt < 4; ++nt) {
        V16 b;
        b.h[0] = *(const v8h*)(base + (nt * 2) * 256);
        b.h[1] = *(const v8h*)(base + (nt * 2 + 1) * 256);
        acc[nt] = __builtin_amdgcn_wmma_f32_16x16x32_f16(
            false, a.v, false, b.v, (short)0, acc[nt], false, false);
      }
    }
    #pragma unroll 1
    for (int kbb = 0; kbb < II / 32; ++kbb) {
      const int kb = kbb * 32;
      V16 a;
      a.h[0] = *(const v8h*)&sh_x[mA][kb + koff];
      a.h[1] = *(const v8h*)&sh_x[mA][kb + 16 + koff];
      const _Float16* base = wih + (kbb * 8 + wave) * BLK + lane8;
      #pragma unroll
      for (int nt = 0; nt < 4; ++nt) {
        V16 b;
        b.h[0] = *(const v8h*)(base + (nt * 2) * 256);
        b.h[1] = *(const v8h*)(base + (nt * 2 + 1) * 256);
        acc[nt] = __builtin_amdgcn_wmma_f32_16x16x32_f16(
            false, a.v, false, b.v, (short)0, acc[nt], false, false);
      }
    }
    __syncthreads();  // protect sh_r readers before we overwrite it

    // ---- phase 3: Euler update of h; r = tanh(h_new) via v_tanh_f32 ----
    #pragma unroll
    for (int nt = 0; nt < 4; ++nt) {
      const int n  = wave * 64 + nt * 16 + nlane;
      const float bh = bhh[n];
      #pragma unroll
      for (int v = 0; v < 8; ++v) {
        const int M = mrow + v;
        const float hold = sh_h[M][n];
        const int g = ((bbase + M) * TT + t) * HH + n;
        const float hnew =
            hold + K_ALPHA * (-hold + acc[nt][v] + bh + K_SIGMA_RE * re_noise[g]);
        sh_h[M][n] = hnew;
        sh_r[M][n] = (_Float16)htanh(hnew);
      }
    }
    __syncthreads();  // sh_r (new) ready for output GEMM + next step

    // ---- phase 4: out = tanh(h_new) @ Who^T + bho (waves 0-3) ----
    if (wave < 4) {
      v8f oc = (v8f){0,0,0,0,0,0,0,0};
      #pragma unroll 1
      for (int kbb = 0; kbb < HH / 32; ++kbb) {
        const int kb = kbb * 32;
        V16 a, b;
        a.h[0] = *(const v8h*)&sh_r[mA][kb + koff];
        a.h[1] = *(const v8h*)&sh_r[mA][kb + 16 + koff];
        const _Float16* base = who + kbb * BLK + lane8;
        b.h[0] = *(const v8h*)(base + (wave * 2) * 256);
        b.h[1] = *(const v8h*)(base + (wave * 2 + 1) * 256);
        oc = __builtin_amdgcn_wmma_f32_16x16x32_f16(
            false, a.v, false, b.v, (short)0, oc, false, false);
      }
      const int o = wave * 16 + nlane;
      const float bo = bho[o];
      #pragma unroll
      for (int v = 0; v < 8; ++v) {
        const int M = mrow + v;
        outs[((bbase + M) * TT + t) * OO + o] = oc[v] + bo;
      }
    }
    // No barrier needed: next phase 1 touches only sh_x.
  }

  __syncthreads();
  for (int e = tid; e < 16 * HH; e += 256) {
    int m = e >> 9, n = e & (HH - 1);
    hfinal[(bbase + m) * HH + n] = sh_h[m][n];
  }
}

// ---------------------------------------------------------------------------
extern "C" void kernel_launch(void* const* d_in, const int* in_sizes, int n_in,
                              void* d_out, int out_size, void* d_ws, size_t ws_size,
                              hipStream_t stream) {
  (void)in_sizes; (void)n_in; (void)out_size; (void)ws_size;
  const float* inputs      = (const float*)d_in[0];
  const float* hidden      = (const float*)d_in[1];
  const float* input_noise = (const float*)d_in[2];
  const float* re_noise    = (const float*)d_in[3];
  const float* Wih         = (const float*)d_in[4];
  const float* Whh         = (const float*)d_in[5];
  const float* bhh         = (const float*)d_in[6];
  const float* Who         = (const float*)d_in[7];
  const float* bho         = (const float*)d_in[8];

  // Workspace layout: WhhTl (512 KB) | WihTl (128 KB) | WhoTl (64 KB)
  char* ws = (char*)d_ws;
  _Float16* WhhTl = (_Float16*)(ws);
  _Float16* WihTl = (_Float16*)(ws + (size_t)HH * HH * sizeof(_Float16));
  _Float16* WhoTl = (_Float16*)(ws + (size_t)HH * HH * sizeof(_Float16)
                                   + (size_t)II * HH * sizeof(_Float16));

  float* out    = (float*)d_out;
  float* hfinal = out;                       // tuple element 0: [B,H]
  float* outs   = out + (size_t)BB * HH;     // tuple element 1: [B,T,O]

  rnn_prep_weights<<<512, 256, 0, stream>>>(Wih, Whh, Who, WihTl, WhhTl, WhoTl);
  rnn_persistent<<<BB / 16, 256, 0, stream>>>(inputs, hidden, input_noise,
                                              re_noise, WhhTl, WihTl, WhoTl,
                                              bhh, bho, hfinal, outs);
}
